// DecoderRNN_21010980012317
// MI455X (gfx1250) — matrix operand order repaired and checked by reading, not compiled, not run on hardware
//
#include <hip/hip_runtime.h>
#include <hip/hip_bf16.h>
#include <math.h>

typedef __bf16 bf16_t;
typedef __attribute__((ext_vector_type(16))) __bf16 v16bf;
typedef __attribute__((ext_vector_type(8)))  float  v8f;

#define B_  64
#define H_  1024
#define V_  50257
#define T_  7
#define G3H (3 * H_)

// ---------------- fp32 -> bf16 weight conversion (one-time) ----------------
__global__ void cvt_f32_bf16(const float* __restrict__ src, bf16_t* __restrict__ dst, long n) {
    long i = (long)blockIdx.x * blockDim.x + threadIdx.x;
    long stride = (long)gridDim.x * blockDim.x;
    for (; i < n; i += stride) dst[i] = (bf16_t)src[i];
}

// ---------------- init: h = encoder_op[0], x = target[:,0] ----------------
__global__ void init_state(const float* __restrict__ enc, const int* __restrict__ target,
                           float* __restrict__ h, bf16_t* __restrict__ hbf, int* __restrict__ x) {
    int i = blockIdx.x * blockDim.x + threadIdx.x;
    if (i < B_ * H_) { float v = enc[i]; h[i] = v; hbf[i] = (bf16_t)v; }
    if (i < B_) x[i] = target[i * T_];
}

// ---------------- e = bf16(relu(emb[x])) ----------------
__global__ void embed_relu(const float* __restrict__ emb, const int* __restrict__ x,
                           bf16_t* __restrict__ e) {
    int i = blockIdx.x * blockDim.x + threadIdx.x;
    if (i >= B_ * H_) return;
    int b = i >> 10, j = i & (H_ - 1);
    float v = emb[(long)x[b] * H_ + j];
    e[i] = (bf16_t)fmaxf(v, 0.0f);
}

// ---------------- WMMA bf16 GEMM: out[m,n] = sum_k A[m,k]*W[n,k] + bias[n] ----------------
// A: 64 x K bf16 row-major. W: N x K bf16 row-major (i.e. B = W^T, column-major in memory,
// so each lane's B-fragment is 16 contiguous bf16 = one 32B read). fp32 accumulate.
// Block = 128 threads = 4 waves; wave w owns m-tile w; grid.x = ceil(N/16) n-tiles.
__global__ __launch_bounds__(128)
void wmma_gemm_bias(const bf16_t* __restrict__ A, const bf16_t* __restrict__ W,
                    const float* __restrict__ bias, float* __restrict__ out,
                    int N, int K, long ldOut) {
    const int lane  = threadIdx.x & 31;
    const int mtile = threadIdx.x >> 5;
    const int n0    = blockIdx.x * 16;

    // A fragment addressing (16x32 bf16 layout): lane<16 -> K offsets 0..15,
    // lane>=16 -> 16..31; row = mtile*16 + lane%16.
    const int kbA = (lane < 16) ? 0 : 16;
    // B fragment (32x16 bf16 layout): lane<16 -> K 0..15, lane>=16 -> K 16..31; col = n0 + lane%16.
    const int kbB = (lane < 16) ? 0 : 16;

    const int row  = mtile * 16 + (lane & 15);
    int col        = n0 + (lane & 15);
    const int colc = (col < N) ? col : (N - 1);   // clamp loads; stores are guarded

    const bf16_t* __restrict__ arow = A + (long)row  * K;
    const bf16_t* __restrict__ brow = W + (long)colc * K;

    v8f c = {0.f, 0.f, 0.f, 0.f, 0.f, 0.f, 0.f, 0.f};

    for (int ks = 0; ks < K; ks += 32) {
        union { v16bf v; uint4 u[2]; } a, b;
        a.u[0] = *(const uint4*)(arow + ks + kbA);        // K = ks + kbA   .. +7
        a.u[1] = *(const uint4*)(arow + ks + kbA + 8);    // K = ks + kbA+8 .. +15
        b.u[0] = *(const uint4*)(brow + ks + kbB);        // K = ks + kbB   .. +7
        b.u[1] = *(const uint4*)(brow + ks + kbB + 8);    // K = ks + kbB+8 .. +15
        __builtin_prefetch(brow + ks + 256, 0, 0);        // stream the weight rows
        c = __builtin_amdgcn_wmma_f32_16x16x32_bf16(
                /*neg_a=*/false, a.v, /*neg_b=*/false, b.v,
                /*c_mod=*/(short)0, c, /*reuse_a=*/false, /*reuse_b=*/false);
    }

    // C layout: VGPR r -> M = r (lanes 0-15) / r+8 (lanes 16-31), N = lane%16.
    const int rowS = mtile * 16 + (lane >> 4) * 8;
    const int colS = n0 + (lane & 15);
    if (colS < N) {
        float bb = bias[colS];
#pragma unroll
        for (int r = 0; r < 8; ++r)
            out[(long)(rowS + r) * ldOut + colS] = c[r] + bb;
    }
}

// ---------------- GRU gate math ----------------
__global__ void gru_gates(const float* __restrict__ gi, const float* __restrict__ gh,
                          float* __restrict__ h, bf16_t* __restrict__ hbf) {
    int i = blockIdx.x * blockDim.x + threadIdx.x;
    if (i >= B_ * H_) return;
    int m = i >> 10, j = i & (H_ - 1);
    const float* gim = gi + (long)m * G3H;
    const float* ghm = gh + (long)m * G3H;
    float r = 1.f / (1.f + __expf(-(gim[j]          + ghm[j])));
    float z = 1.f / (1.f + __expf(-(gim[H_ + j]     + ghm[H_ + j])));
    float n = tanhf(gim[2 * H_ + j] + r * ghm[2 * H_ + j]);
    float hn = (1.f - z) * n + z * h[i];
    h[i] = hn;
    hbf[i] = (bf16_t)hn;
}

// ---------------- argmax over V (first-max tie-break, like jnp.argmax) ----------------
__global__ void argmax_row(const float* __restrict__ logits, long rowStride, int* __restrict__ x) {
    const int b = blockIdx.x;
    const float* row = logits + (long)b * rowStride;
    __shared__ float smax[256];
    __shared__ int   sidx[256];
    float best = -INFINITY; int bi = 0;
    for (int j = threadIdx.x; j < V_; j += blockDim.x) {
        float v = row[j];
        if (v > best) { best = v; bi = j; }
    }
    smax[threadIdx.x] = best; sidx[threadIdx.x] = bi;
    __syncthreads();
    for (int s = blockDim.x / 2; s > 0; s >>= 1) {
        if ((int)threadIdx.x < s) {
            float v2 = smax[threadIdx.x + s]; int i2 = sidx[threadIdx.x + s];
            float v1 = smax[threadIdx.x];     int i1 = sidx[threadIdx.x];
            if (v2 > v1 || (v2 == v1 && i2 < i1)) { smax[threadIdx.x] = v2; sidx[threadIdx.x] = i2; }
        }
        __syncthreads();
    }
    if (threadIdx.x == 0) x[b] = sidx[0];
}

// ---------------- log_softmax over each (b,t) row ----------------
__global__ void log_softmax_rows(const float* __restrict__ logits, float* __restrict__ lp) {
    long row = blockIdx.x;                    // b*T + t
    const float* in = logits + row * (long)V_;
    float* out      = lp     + row * (long)V_;
    __shared__ float sred[256];
    float mx = -INFINITY;
    for (int j = threadIdx.x; j < V_; j += blockDim.x) mx = fmaxf(mx, in[j]);
    sred[threadIdx.x] = mx; __syncthreads();
    for (int s = 128; s > 0; s >>= 1) {
        if ((int)threadIdx.x < s) sred[threadIdx.x] = fmaxf(sred[threadIdx.x], sred[threadIdx.x + s]);
        __syncthreads();
    }
    mx = sred[0]; __syncthreads();
    float sum = 0.f;
    for (int j = threadIdx.x; j < V_; j += blockDim.x) sum += __expf(in[j] - mx);
    sred[threadIdx.x] = sum; __syncthreads();
    for (int s = 128; s > 0; s >>= 1) {
        if ((int)threadIdx.x < s) sred[threadIdx.x] += sred[threadIdx.x + s];
        __syncthreads();
    }
    float lse = mx + __logf(sred[0]);
    for (int j = threadIdx.x; j < V_; j += blockDim.x) out[j] = in[j] - lse;
}

// ---------------- host orchestration ----------------
extern "C" void kernel_launch(void* const* d_in, const int* in_sizes, int n_in,
                              void* d_out, int out_size, void* d_ws, size_t ws_size,
                              hipStream_t stream) {
    (void)in_sizes; (void)n_in; (void)out_size; (void)ws_size;
    const int*   target = (const int*)  d_in[0];
    const float* enc    = (const float*)d_in[1];
    const float* emb    = (const float*)d_in[2];
    const float* w_ih   = (const float*)d_in[3];
    const float* w_hh   = (const float*)d_in[4];
    const float* b_ih   = (const float*)d_in[5];
    const float* b_hh   = (const float*)d_in[6];
    const float* lin_w  = (const float*)d_in[7];
    const float* lin_b  = (const float*)d_in[8];

    float* log_probs = (float*)d_out;                         // (B, T, V)
    float* logits    = (float*)d_out + (long)B_ * T_ * V_;    // (B, T, V)

    // workspace carve-up
    char* ws = (char*)d_ws;
    auto take = [&](size_t bytes) { char* p = ws; ws += (bytes + 255) & ~(size_t)255; return p; };
    bf16_t* linw_bf = (bf16_t*)take((size_t)V_ * H_ * sizeof(bf16_t));    // 103 MB
    bf16_t* wih_bf  = (bf16_t*)take((size_t)G3H * H_ * sizeof(bf16_t));   // 6.3 MB
    bf16_t* whh_bf  = (bf16_t*)take((size_t)G3H * H_ * sizeof(bf16_t));   // 6.3 MB
    float*  h       = (float*) take((size_t)B_ * H_ * sizeof(float));
    bf16_t* hbf     = (bf16_t*)take((size_t)B_ * H_ * sizeof(bf16_t));
    bf16_t* ebf     = (bf16_t*)take((size_t)B_ * H_ * sizeof(bf16_t));
    float*  gi      = (float*) take((size_t)B_ * G3H * sizeof(float));
    float*  gh      = (float*) take((size_t)B_ * G3H * sizeof(float));
    int*    x       = (int*)   take((size_t)B_ * sizeof(int));

    // one-time bf16 weight conversion (halves the dominant HBM stream)
    cvt_f32_bf16<<<2048, 256, 0, stream>>>(lin_w, linw_bf, (long)V_ * H_);
    cvt_f32_bf16<<<512,  256, 0, stream>>>(w_ih,  wih_bf,  (long)G3H * H_);
    cvt_f32_bf16<<<512,  256, 0, stream>>>(w_hh,  whh_bf,  (long)G3H * H_);

    init_state<<<(B_ * H_ + 255) / 256, 256, 0, stream>>>(enc, target, h, hbf, x);

    for (int t = 0; t < T_; ++t) {
        embed_relu<<<(B_ * H_ + 255) / 256, 256, 0, stream>>>(emb, x, ebf);
        // gi = e @ w_ih^T + b_ih ; gh = h @ w_hh^T + b_hh   (N = 3072, K = 1024)
        wmma_gemm_bias<<<G3H / 16, 128, 0, stream>>>(ebf, wih_bf, b_ih, gi, G3H, H_, (long)G3H);
        wmma_gemm_bias<<<G3H / 16, 128, 0, stream>>>(hbf, whh_bf, b_hh, gh, G3H, H_, (long)G3H);
        gru_gates<<<(B_ * H_ + 255) / 256, 256, 0, stream>>>(gi, gh, h, hbf);
        // logits[:, t, :] = h @ lin_w^T + lin_b  (N = 50257)
        wmma_gemm_bias<<<(V_ + 15) / 16, 128, 0, stream>>>(
            hbf, linw_bf, lin_b, logits + (long)t * V_, V_, H_, (long)T_ * V_);
        argmax_row<<<B_, 256, 0, stream>>>(logits + (long)t * V_, (long)T_ * V_, x);
    }

    log_softmax_rows<<<B_ * T_, 256, 0, stream>>>(logits, log_probs);
}
